// accident_46858093199669
// MI455X (gfx1250) — matrix operand order repaired and checked by reading, not compiled, not run on hardware
//
#include <hip/hip_runtime.h>

// Problem constants from the reference
#define NB 4
#define CD 64          // C_IN == C_OUT == 64
#define VD 8192
#define KSTEP 32       // WMMA bf16 K
#define WTILE 64       // w columns per block (4 waves x 16)
#define NSTEPS (VD / KSTEP)            // 256
#define OUT_A_OFF ((size_t)NB * CD * VD) // elems of `out` before the A copy

typedef __attribute__((ext_vector_type(16))) __bf16 v16bf;
typedef __attribute__((ext_vector_type(8)))  float  v8f;
typedef __attribute__((ext_vector_type(4)))  float  v4f;   // clang-native float4
typedef __attribute__((ext_vector_type(4)))  unsigned int v4u;

union Frag16 { v16bf v; v4u q[2]; };

// ---------------------------------------------------------------------------
// Kernel 1: y[n,c,v] = sum_c' W[c,c'] * x[n,c',v] + b[c]   (stored as bf16)
// ---------------------------------------------------------------------------
__global__ __launch_bounds__(256) void yproj_kernel(const float* __restrict__ x,
                                                    const float* __restrict__ W,
                                                    const float* __restrict__ bias,
                                                    __bf16* __restrict__ y) {
    __shared__ float Ws[CD * CD];
    __shared__ float bs[CD];
    const int tid = threadIdx.x;
    const int n   = blockIdx.y;
    const int v   = blockIdx.x * 256 + tid;

    for (int i = tid; i < CD * CD; i += 256) Ws[i] = W[i];
    if (tid < CD) bs[tid] = bias[tid];
    __syncthreads();

    float acc[CD];
#pragma unroll
    for (int c = 0; c < CD; ++c) acc[c] = bs[c];

    for (int cp = 0; cp < CD; ++cp) {
        const float xv = x[((size_t)n * CD + cp) * VD + v];
#pragma unroll
        for (int c = 0; c < CD; ++c)
            acc[c] = fmaf(Ws[c * CD + cp], xv, acc[c]);
    }
#pragma unroll
    for (int c = 0; c < CD; ++c)
        y[((size_t)n * CD + c) * VD + v] = (__bf16)acc[c];
}

// ---------------------------------------------------------------------------
// Kernel 2: out[n,c,w] = sum_v y[n,c,v] * A[n,v,w]  via v_wmma_f32_16x16x32_bf16
//           + fused streaming copy of A (f32) into d_out's second region.
// ---------------------------------------------------------------------------
__global__ __launch_bounds__(128) void bigmm_kernel(const float* __restrict__ A,
                                                    const __bf16* __restrict__ y,
                                                    float* __restrict__ out) {
    // A tile staged TRANSPOSED: aT[w][k], stride 48 bf16 = 96 B (16B aligned rows)
    __shared__ __bf16 aT[WTILE][48];
    // y tile staged row-major: yS[c][k]
    __shared__ __bf16 yS[CD][KSTEP];

    const int tid  = threadIdx.x;
    const int lane = tid & 31;
    const int wid  = tid >> 5;                 // wave id 0..3 -> w sub-tile
    const int n    = blockIdx.y;
    const int w0   = blockIdx.x * WTILE;

    const float*  Ab  = A   + (size_t)n * VD * VD;
    float*        Acp = out + OUT_A_OFF + (size_t)n * VD * VD;
    const __bf16* yb  = y   + (size_t)n * CD * VD;

    const int lhalf = (lane < 16) ? 0 : 1;
    const int l15   = lane & 15;

    // Per-lane LDS fragment bases (constant across K loop):
    // B operand (A data), 32x16 bf16: lanes 0-15 K=0..15, lanes 16-31 K=16..31,
    // two K packed per VGPR -> two aligned b128 chunks at kb and kb+8.
    const __bf16* bsrc = &aT[16 * wid + l15][lhalf * 16];

    v8f acc[4] = {};   // 4 c-tiles x (16x16 f32)

    for (int s = 0; s < NSTEPS; ++s) {
        const int v0 = s * KSTEP;

        // ---- coalesced global loads: A tile (32 x 64 f32) as b128 ----
        v4f a4[4];
        size_t gidx[4];
#pragma unroll
        for (int j = 0; j < 4; ++j) {
            const int fi = tid + 128 * j;      // 0..511 float4 chunks
            const int k  = fi >> 4;            // 0..31 (v row)
            const int c4 = fi & 15;            // float4 column group
            gidx[j] = (size_t)(v0 + k) * VD + (size_t)w0 + (size_t)c4 * 4;
            a4[j] = *reinterpret_cast<const v4f*>(Ab + gidx[j]);
        }
        // ---- y tile (64 x 32 bf16) as b128 ----
        v4u y4[2];
#pragma unroll
        for (int j = 0; j < 2; ++j) {
            const int ci   = tid + 128 * j;    // 0..255 16B chunks
            const int row  = ci >> 2;
            const int part = ci & 3;
            y4[j] = *reinterpret_cast<const v4u*>(yb + (size_t)row * VD + v0 + part * 8);
        }
        // Prefetch A two K-steps ahead (global_prefetch_b8)
        if (s + 2 < NSTEPS)
            __builtin_prefetch(Ab + gidx[0] + (size_t)2 * KSTEP * VD, 0, 0);

        // ---- fused A copy into d_out (streaming, nontemporal) ----
#pragma unroll
        for (int j = 0; j < 4; ++j)
            __builtin_nontemporal_store(a4[j], reinterpret_cast<v4f*>(Acp + gidx[j]));

        __syncthreads();   // previous iteration's LDS fragment reads complete

        // ---- stage A (cvt f32->bf16, transposed) and y into LDS ----
#pragma unroll
        for (int j = 0; j < 4; ++j) {
            const int fi = tid + 128 * j;
            const int k  = fi >> 4;
            const int c4 = fi & 15;
            aT[c4 * 4 + 0][k] = (__bf16)a4[j].x;
            aT[c4 * 4 + 1][k] = (__bf16)a4[j].y;
            aT[c4 * 4 + 2][k] = (__bf16)a4[j].z;
            aT[c4 * 4 + 3][k] = (__bf16)a4[j].w;
        }
#pragma unroll
        for (int j = 0; j < 2; ++j) {
            const int ci   = tid + 128 * j;
            const int row  = ci >> 2;
            const int part = ci & 3;
            *reinterpret_cast<v4u*>(&yS[row][part * 8]) = y4[j];
        }
        __syncthreads();

        // ---- fragments + 4 WMMAs per wave ----
        Frag16 bfrag;
        bfrag.q[0] = *reinterpret_cast<const v4u*>(bsrc);
        bfrag.q[1] = *reinterpret_cast<const v4u*>(bsrc + 8);

#pragma unroll
        for (int m = 0; m < 4; ++m) {
            // A operand (y data), 16x32 bf16: lanes 0-15 K=0..15,
            // lanes 16-31 K=16..31 -> two aligned b128 chunks.
            Frag16 afrag;
            const __bf16* asrc = &yS[16 * m + l15][lhalf * 16];
            afrag.q[0] = *reinterpret_cast<const v4u*>(asrc);
            afrag.q[1] = *reinterpret_cast<const v4u*>(asrc + 8);
            acc[m] = __builtin_amdgcn_wmma_f32_16x16x32_bf16(
                /*neg_a=*/false, afrag.v, /*neg_b=*/false, bfrag.v,
                /*c_mod=*/(short)0, acc[m], /*reuse_a=*/false, /*reuse_b=*/false);
        }
    }

    // ---- epilogue: C/D layout -> out[n,c,w] ----
    // VGPR r, lanes 0-15: M=r ; lanes 16-31: M=r+8 ; N = lane&15
    const int wcol   = w0 + 16 * wid + l15;
    const int rowoff = lhalf * 8;
#pragma unroll
    for (int m = 0; m < 4; ++m) {
#pragma unroll
        for (int r = 0; r < 8; ++r) {
            const int c = 16 * m + r + rowoff;
            out[((size_t)n * CD + c) * VD + wcol] = acc[m][r];
        }
    }
}

// ---------------------------------------------------------------------------
extern "C" void kernel_launch(void* const* d_in, const int* in_sizes, int n_in,
                              void* d_out, int out_size, void* d_ws, size_t ws_size,
                              hipStream_t stream) {
    const float* x = (const float*)d_in[0];   // (N, C_IN, 1, V)
    const float* A = (const float*)d_in[1];   // (N, V, V)
    const float* W = (const float*)d_in[2];   // (C_OUT, C_IN, 1, 1)
    const float* b = (const float*)d_in[3];   // (C_OUT,)
    float* out = (float*)d_out;               // [out | A-copy] concatenated
    __bf16* y = (__bf16*)d_ws;                // needs NB*CD*VD*2 = 4 MB scratch

    dim3 g1(VD / 256, NB);
    yproj_kernel<<<g1, 256, 0, stream>>>(x, W, b, y);

    dim3 g2(VD / WTILE, NB);
    bigmm_kernel<<<g2, 128, 0, stream>>>(A, y, out);
}
